// LocationSensitiveAttention_21835613733413
// MI455X (gfx1250) — matrix operand order repaired and checked by reading, not compiled, not run on hardware
//
#include <hip/hip_runtime.h>
#include <hip/hip_bf16.h>

namespace {

constexpr int BATCH = 64;
constexpr int TLEN  = 1024;
constexpr int ADIM  = 128;
constexpr int EDIM  = 512;
constexpr int NFILT = 32;
constexpr int KSZ   = 31;
constexpr int PADK  = 15;        // (K-1)/2
constexpr int TT    = 128;       // t-rows handled per workgroup in energies kernel
constexpr int TILES = TT / 16;   // 8 WMMA t-tiles
constexpr int CATW  = 160;       // staged attention_weights_cat window (158 used, padded)

// ---- WMMA types (CDNA5 gfx1250, wave32) ----
typedef __attribute__((ext_vector_type(16))) __bf16         v16bf;
typedef __attribute__((ext_vector_type(8)))  float          v8f;
typedef __attribute__((ext_vector_type(8)))  unsigned short us8;

union Frag { v16bf f; us8 h[2]; };

__device__ inline unsigned short f2bf(float x) {
  unsigned int u = __float_as_uint(x);
  u += 0x7FFFu + ((u >> 16) & 1u);        // round-to-nearest-even f32 -> bf16
  return (unsigned short)(u >> 16);
}
__device__ inline ushort4 f2bf4(float4 x) {
  ushort4 r;
  r.x = f2bf(x.x); r.y = f2bf(x.y); r.z = f2bf(x.z); r.w = f2bf(x.w);
  return r;
}

// Branch-free tanh: (e^2x - 1) / (e^2x + 1), input clamped so exp never
// overflows (tanh(+-10) == +-1 to fp32 precision). One v_exp_f32 + v_rcp_f32,
// no EXEC-mask branching (unlike the OCML tanhf expansion).
__device__ inline float fast_tanh(float x) {
  float xc = fminf(fmaxf(x, -10.f), 10.f);
  float ex = __expf(2.f * xc);
  return (ex - 1.f) * __frcp_rn(ex + 1.f);
}

// Dynamic-LDS layout for the energies kernel (bytes)
constexpr unsigned SM_WM_OFF  = 0;                                   // Wm as bf16: 128*512*2 = 131072
constexpr unsigned SM_A_OFF   = SM_WM_OFF + ADIM * EDIM * 2;         // A tile bf16: 16*512*2 = 16384
constexpr unsigned SM_CAT_OFF = SM_A_OFF + 16 * EDIM * 2;            // cat window: 2*160 floats
constexpr unsigned SM_E_OFF   = SM_CAT_OFF + 2 * CATW * 4;           // 16-entry energy accumulator
constexpr unsigned SMEM_TOTAL = SM_E_OFF + 16 * 4;                   // = 148800 bytes

// ============================================================================
// Kernel 1: pq[b][a] = hidden·Wq^T ; Weff[a][c][k] = sum_f Wl[a][f]*conv_w[f][c][k]
// ============================================================================
__global__ __launch_bounds__(256)
void lsa_prep_kernel(const float* __restrict__ hidden, const float* __restrict__ Wq,
                     const float* __restrict__ Wl, const float* __restrict__ conv_w,
                     float* __restrict__ pq, float* __restrict__ weff) {
  int idx = blockIdx.x * 256 + threadIdx.x;
  if (idx < BATCH * ADIM) {
    int b = idx / ADIM, a = idx % ADIM;
    float s = 0.f;
    for (int j = 0; j < ADIM; ++j) s += hidden[b * ADIM + j] * Wq[a * ADIM + j];
    pq[idx] = s;
  } else {
    int i = idx - BATCH * ADIM;
    if (i < ADIM * 2 * KSZ) {
      int a = i / (2 * KSZ);
      int rem = i % (2 * KSZ);
      int c = rem / KSZ, k = rem % KSZ;
      float s = 0.f;
      for (int f = 0; f < NFILT; ++f)
        s += Wl[a * NFILT + f] * conv_w[(f * 2 + c) * KSZ + k];
      weff[i] = s;   // layout: [a][c][k]
    }
  }
}

// ============================================================================
// Kernel 2: energies[b][t] = v · tanh(pq + pm + pa) via bf16 WMMA for pm
//   grid: BATCH * (TLEN/TT) = 512 blocks, 256 threads (8 wave32s)
//   wave w owns a-tile [16w,16w+16); workgroup owns b = blk/8, t0wg = (blk%8)*128
// ============================================================================
__global__ __launch_bounds__(256)
void lsa_energies_kernel(const float* __restrict__ pmem,   // processed_memory [B,T,E]
                         const float* __restrict__ cat,    // attention_weights_cat [B,2,T]
                         const float* __restrict__ Wm,     // [A,E]
                         const float* __restrict__ pq,     // [B,A]
                         const float* __restrict__ vvec,   // [A]
                         const float* __restrict__ weff,   // [A,2,K]
                         float* __restrict__ eout) {       // [B,T]
  extern __shared__ char smem[];
  unsigned short* sWm = (unsigned short*)(smem + SM_WM_OFF);   // [A][E] bf16
  unsigned short* sA  = (unsigned short*)(smem + SM_A_OFF);    // [16][E] bf16
  float*          sCat = (float*)(smem + SM_CAT_OFF);          // [2][CATW]
  float*          sE   = (float*)(smem + SM_E_OFF);            // [16]

  const int tid  = threadIdx.x;
  const int wave = tid >> 5;
  const int lane = tid & 31;
  const int b    = blockIdx.x >> 3;
  const int t0wg = (blockIdx.x & 7) * TT;

  // ---- Stage Wm (fp32 from L2, 256 KB) into LDS as bf16 (128 KB), once ----
  {
    const float4* wm4 = (const float4*)Wm;
    ushort4* d4 = (ushort4*)sWm;
    for (int i = tid; i < ADIM * EDIM / 4; i += 256) d4[i] = f2bf4(wm4[i]);
  }
  // ---- Stage attention_weights_cat window for the 128 rows (+/- PADK) ----
  for (int i = tid; i < 2 * CATW; i += 256) {
    int c = i / CATW, j = i % CATW;
    int t = t0wg - PADK + j;
    float val = (j < TT + 2 * PADK && t >= 0 && t < TLEN)
                    ? cat[((size_t)b * 2 + c) * TLEN + t] : 0.f;
    sCat[c * CATW + j] = val;
  }
  __syncthreads();

  const int n  = lane & 15;     // N index within a-tile (and A-fragment row)
  const int hh = lane >> 4;     // lane half
  const int a  = wave * 16 + n; // global a for this lane's output column
  const float vq = pq[b * ADIM + a];
  const float va = vvec[a];

  for (int tile = 0; tile < TILES; ++tile) {
    const int t0 = t0wg + tile * 16;
    if (tid < 16) sE[tid] = 0.f;

    // ---- Stage 16x512 fp32 rows of processed_memory -> bf16 LDS ----
    {
      const float4* pm4 = (const float4*)(pmem + ((size_t)b * TLEN + t0) * EDIM);
      ushort4* d4 = (ushort4*)sA;
      for (int i = tid; i < 16 * EDIM / 4; i += 256) d4[i] = f2bf4(pm4[i]);
    }
    // Prefetch next tile while we compute (global_prefetch_b8)
    if (tile + 1 < TILES) {
      const char* nxt = (const char*)(pmem + ((size_t)b * TLEN + t0 + 16) * EDIM);
      __builtin_prefetch(nxt + tid * 128, 0, 1);
    }
    __syncthreads();

    // ---- pm tile via v_wmma_f32_16x16x32_bf16, K=512 -> 16 chained WMMAs.
    //      unroll capped at 4 so only 4 fragment pairs stay live (VGPR budget:
    //      keeps the wave under 256 VGPRs -> more resident waves per SIMD). ----
    v8f acc = {};
#pragma unroll 4
    for (int kb = 0; kb < EDIM; kb += 32) {
      Frag fa, fb;
      // A (16x32 bf16): lane row m=lane%16; elems 0..7 -> k=kb+8*hh+0..7,
      //                 elems 8..15 -> k=kb+16+8*hh+0..7  (ISA 7.12.2)
      fa.h[0] = *(const us8*)&sA[n * EDIM + kb + 8 * hh];
      fa.h[1] = *(const us8*)&sA[n * EDIM + kb + 16 + 8 * hh];
      // B (32x16 bf16): lane col n=lane%16; elems 0..7 -> k=kb+8*hh+0..7,
      //                 elems 8..15 -> k=kb+16+8*hh+0..7 (same as A)
      fb.h[0] = *(const us8*)&sWm[a * EDIM + kb + 8 * hh];
      fb.h[1] = *(const us8*)&sWm[a * EDIM + kb + 16 + 8 * hh];
      acc = __builtin_amdgcn_wmma_f32_16x16x32_bf16(
          /*neg_a=*/false, fa.f, /*neg_b=*/false, fb.f,
          /*c_mod=*/(short)0, acc, /*reuse_a=*/false, /*reuse_b=*/false);
    }

    // ---- pa via folded 62-tap filter; rows for this lane: t = t0 + r + 8*hh ----
    float pa[8];
#pragma unroll
    for (int r = 0; r < 8; ++r) pa[r] = 0.f;
    const int ltbase = tile * 16 + 8 * hh;
    for (int c = 0; c < 2; ++c) {
      const float* wrow = weff + ((size_t)a * 2 + c) * KSZ;
      const float* crow = sCat + c * CATW;
      for (int k = 0; k < KSZ; ++k) {
        float wv = wrow[k];
#pragma unroll
        for (int r = 0; r < 8; ++r) pa[r] += wv * crow[ltbase + r + k];
      }
    }

    // ---- energy contribution + reduction over A ----
#pragma unroll
    for (int r = 0; r < 8; ++r) {
      float s = va * fast_tanh(acc[r] + vq + pa[r]);
      s += __shfl_xor(s, 1);
      s += __shfl_xor(s, 2);
      s += __shfl_xor(s, 4);
      s += __shfl_xor(s, 8);        // sum over 16 n-lanes (stays within half)
      if (n == 0) atomicAdd(&sE[r + 8 * hh], s);  // ds_add_f32 across 8 waves
    }
    __syncthreads();
    if (tid < 16) eout[(size_t)b * TLEN + t0 + tid] = sE[tid];
    __syncthreads();
  }
}

// ============================================================================
// Kernel 3: masked softmax over T per batch row
// ============================================================================
__global__ __launch_bounds__(256)
void lsa_softmax_kernel(const float* __restrict__ e, const unsigned char* __restrict__ mask,
                        float* __restrict__ wout) {
  __shared__ float red[8];
  const int b = blockIdx.x, tid = threadIdx.x;
  const int wave = tid >> 5, lane = tid & 31;
  const float NEG_INF = __int_as_float(0xff800000);

  float vals[4];
  float mx = NEG_INF;
#pragma unroll
  for (int i = 0; i < 4; ++i) {
    int t = tid + i * 256;
    float x = e[(size_t)b * TLEN + t];
    if (mask[(size_t)b * TLEN + t]) x = NEG_INF;
    vals[i] = x;
    mx = fmaxf(mx, x);
  }
  for (int off = 16; off; off >>= 1) mx = fmaxf(mx, __shfl_xor(mx, off));
  if (lane == 0) red[wave] = mx;
  __syncthreads();
  float m = red[0];
  for (int w = 1; w < 8; ++w) m = fmaxf(m, red[w]);
  __syncthreads();

  float s = 0.f;
#pragma unroll
  for (int i = 0; i < 4; ++i) {
    float p = expf(vals[i] - m);
    vals[i] = p;
    s += p;
  }
  for (int off = 16; off; off >>= 1) s += __shfl_xor(s, off);
  if (lane == 0) red[wave] = s;
  __syncthreads();
  float tot = 0.f;
  for (int w = 0; w < 8; ++w) tot += red[w];
  float inv = 1.f / tot;
#pragma unroll
  for (int i = 0; i < 4; ++i) wout[(size_t)b * TLEN + tid + i * 256] = vals[i] * inv;
}

// ============================================================================
// Kernel 4: context[b][e] = sum_t w[b][t] * memory[b][t][e]
//   grid: 128 blocks (2 per batch row), 256 threads, coalesced over e
// ============================================================================
__global__ __launch_bounds__(256)
void lsa_context_kernel(const float* __restrict__ w, const float* __restrict__ memory,
                        float* __restrict__ out) {
  __shared__ float sw[256];
  const int b = blockIdx.x >> 1;
  const int e = ((blockIdx.x & 1) << 8) + threadIdx.x;
  float accum = 0.f;
  for (int t0 = 0; t0 < TLEN; t0 += 256) {
    __syncthreads();
    sw[threadIdx.x] = w[(size_t)b * TLEN + t0 + threadIdx.x];
    __syncthreads();
    const float* mrow = memory + ((size_t)b * TLEN + t0) * EDIM + e;
    for (int j = 0; j < 256; ++j) accum += sw[j] * mrow[(size_t)j * EDIM];
  }
  out[(size_t)b * EDIM + e] = accum;
}

} // namespace

// ============================================================================
extern "C" void kernel_launch(void* const* d_in, const int* in_sizes, int n_in,
                              void* d_out, int out_size, void* d_ws, size_t ws_size,
                              hipStream_t stream) {
  const float*         hidden = (const float*)d_in[0];  // [B,A]
  const float*         memory = (const float*)d_in[1];  // [B,T,E]
  const float*         pmem   = (const float*)d_in[2];  // [B,T,E]
  const float*         cat    = (const float*)d_in[3];  // [B,2,T]
  const unsigned char* mask   = (const unsigned char*)d_in[4];  // [B,T] bool
  const float*         Wq     = (const float*)d_in[5];  // [A,A]
  const float*         Wm     = (const float*)d_in[6];  // [A,E]
  const float*         vvec   = (const float*)d_in[7];  // [1,A]
  const float*         conv_w = (const float*)d_in[8];  // [NF,2,K]
  const float*         Wl     = (const float*)d_in[9];  // [A,NF]

  float* ws   = (float*)d_ws;
  float* pq   = ws;                 // B*A      = 8192
  float* weff = ws + 8192;          // A*2*K    = 7936
  float* e    = ws + 16384;         // B*T      = 65536

  float* out_ctx = (float*)d_out;                    // [B,E]
  float* out_w   = (float*)d_out + BATCH * EDIM;     // [B,T]

  lsa_prep_kernel<<<63, 256, 0, stream>>>(hidden, Wq, Wl, conv_w, pq, weff);
  lsa_energies_kernel<<<BATCH * (TLEN / TT), 256, SMEM_TOTAL, stream>>>(
      pmem, cat, Wm, pq, vvec, weff, e);
  lsa_softmax_kernel<<<BATCH, 256, 0, stream>>>(e, mask, out_w);
  lsa_context_kernel<<<BATCH * 2, 256, 0, stream>>>(out_w, memory, out_ctx);
}
